// Mamba2CrossBlock_63505386439329
// MI455X (gfx1250) — compile-verified, matrix operand order checked by reading, not run on hardware
//
#include <hip/hip_runtime.h>

#define DEV __device__ __forceinline__

typedef __attribute__((ext_vector_type(16))) __bf16 v16bf;
typedef __attribute__((ext_vector_type(8)))  __bf16 v8bf;
typedef __attribute__((ext_vector_type(8)))  float  v8f;
typedef __attribute__((ext_vector_type(4)))  int    v4i;
typedef __attribute__((ext_vector_type(8)))  int    v8i;

// ---------------------------------------------------------------- constants
#define BB   4
#define GG   256
#define LL   2048
#define TT   2304          // G + L
#define BT   9216          // B*T
#define BL   8192          // B*L
#define DM   1024          // D_MODEL
#define DIN  2048          // D_INNER
#define HH   16            // NHEADS
#define PP   128           // HEADDIM
#define NN   64            // D_STATE
#define CONVD 2176         // CONV_DIM
#define DIP  4240          // D_IN_PROJ
#define NCH  9             // chunks = T/256
#define CHK  256           // CHUNK

// ---------------------------------------------------------------- helpers
DEV __bf16 f2bf(float f) {
  unsigned u = __builtin_bit_cast(unsigned, f);
  unsigned r = u + 0x7FFFu + ((u >> 16) & 1u);      // round-to-nearest-even
  unsigned short h = (unsigned short)(r >> 16);
  return __builtin_bit_cast(__bf16, h);
}
DEV float siluf(float x) { return x / (1.0f + expf(-x)); }
DEV float softplusf(float x) { return x > 20.0f ? x : log1pf(expf(x)); }
DEV float geluf(float x) { return 0.5f * x * (1.0f + erff(x * 0.70710678118654752f)); }

DEV v8f wmma_bf16(v16bf a, v16bf b, v8f c) {
  return __builtin_amdgcn_wmma_f32_16x16x32_bf16(false, a, false, b, (short)0, c,
                                                 false, false);
}

// Load a 16x32 bf16 A/B fragment per CDNA5 WMMA VGPR layout.
// lane<16: row=row0+lane, halves = K[k0+0..7], K[k0+16..23]
// lane>=16: row=row0+lane-16, halves = K[k0+8..15], K[k0+24..31]
DEV v16bf load_frag(const __bf16* base, int ld, int row0, int k0) {
  const int lane = threadIdx.x & 31;
  const __bf16* p = base + (size_t)(row0 + (lane & 15)) * ld + k0 + ((lane >> 4) << 3);
  v8bf x0 = *(const v8bf*)p;
  v8bf x1 = *(const v8bf*)(p + 16);
  v16bf a;
#pragma unroll
  for (int i = 0; i < 8; ++i) { a[i] = x0[i]; a[i + 8] = x1[i]; }
  return a;
}

// ---------------------------------------------------------------- TDM (CDNA5)
// Issue a Tensor-Data-Mover DMA of one row of `nelem` f32 from global memory
// into LDS at byte offset lds_byte_off.  D# built per ISA 08_async_tensor §8:
//   group0: count=1 | lds_addr | global_addr[56:0] | type=2
//   group1: data_size=4B, tensor_dim0=tile_dim0=nelem, dims1..=1
// Tracked with TENSORcnt; caller must s_wait_tensorcnt before consuming.
DEV void tdm_load_f32_row(const float* gptr, unsigned lds_byte_off, int nelem) {
  unsigned long long ga = (unsigned long long)gptr;
  v4i g0 = { 1,                                     // count=1, user descriptor
             (int)lds_byte_off,                     // lds_addr
             (int)(unsigned)(ga & 0xFFFFFFFFull),   // global_addr[31:0]
             (int)((unsigned)((ga >> 32) & 0x01FFFFFFull) | 0x80000000u) }; // [56:32]|type=2
  v8i g1 = { 2 << 16,        // data_size = 4 bytes
             nelem << 16,    // tensor_dim0 (low 16b, bits 63:48)
             1 << 16,        // tensor_dim0 hi = 0, tensor_dim1 = 1
             nelem << 16,    // tensor_dim1 hi = 0, tile_dim0 = nelem
             1,              // tile_dim1 = 1, tile_dim2 = 0
             nelem,          // tensor_dim0_stride (low 32b)
             0, 0 };
  asm volatile("tensor_load_to_lds %0, %1" :: "s"(g0), "s"(g1) : "memory");
}

// ---------------------------------------------------------------- f32 -> bf16
__global__ void __launch_bounds__(256)
tobf_k(const float* __restrict__ x, __bf16* __restrict__ o, int n) {
  int i = blockIdx.x * 256 + threadIdx.x;
  if (i < n) o[i] = f2bf(x[i]);
}

// ---------------------------------------------------------------- LayerNorm
__global__ void __launch_bounds__(256)
layernorm_k(const float* __restrict__ x, const float* __restrict__ w,
            const float* __restrict__ bvec, float* __restrict__ outf,
            __bf16* __restrict__ outbf, int seg, int toff) {
  int row = blockIdx.x, tid = threadIdx.x;
  __shared__ float red[256];
  const float* xr = x + (size_t)row * DM;
  float vals[4], s = 0.f;
#pragma unroll
  for (int j = 0; j < 4; ++j) { int d = tid + (j << 8); vals[j] = xr[d]; s += vals[j]; }
  red[tid] = s; __syncthreads();
  for (int off = 128; off; off >>= 1) { if (tid < off) red[tid] += red[tid + off]; __syncthreads(); }
  float mu = red[0] * (1.0f / DM); __syncthreads();
  s = 0.f;
#pragma unroll
  for (int j = 0; j < 4; ++j) { float d = vals[j] - mu; s += d * d; }
  red[tid] = s; __syncthreads();
  for (int off = 128; off; off >>= 1) { if (tid < off) red[tid] += red[tid + off]; __syncthreads(); }
  float inv = rsqrtf(red[0] * (1.0f / DM) + 1e-5f);
  size_t orow = (size_t)((row / seg) * TT + toff + row % seg);
#pragma unroll
  for (int j = 0; j < 4; ++j) {
    int d = tid + (j << 8);
    float y = (vals[j] - mu) * inv * w[d] + bvec[d];
    if (outf) outf[(size_t)row * DM + d] = y;
    outbf[orow * DM + d] = f2bf(y);
  }
}

// ---------------------------------------------------------------- WMMA GEMM
// C[M,N] = A[M,K] (bf16, row-major) x W[N,K]^T (bf16, row-major) [+bias][+GELU]
template <int EPI>   // 0:none 1:bias 2:bias+GELU
__global__ void __launch_bounds__(256)
gemm_bf16_nt(const __bf16* __restrict__ A, const __bf16* __restrict__ W,
             const float* __restrict__ bias, float* __restrict__ Cf,
             __bf16* __restrict__ Cbf, int M, int N, int K) {
  int wid = blockIdx.x * 8 + (threadIdx.x >> 5);
  int mt = M >> 4, nt = (N + 63) >> 6;
  if (wid >= mt * nt) return;
  int m0 = (wid % mt) << 4;
  int n0 = (wid / mt) << 6;
  int lane = threadIdx.x & 31, col = lane & 15, hi = lane >> 4;

  v8f acc[4] = {};
  for (int k0 = 0; k0 < K; k0 += 32) {
    v16bf a = load_frag(A, K, m0, k0);
    if (k0 + 64 < K)  // global_prefetch_b8 on the A stream
      __builtin_prefetch(A + (size_t)(m0 + col) * K + k0 + 64, 0, 3);
#pragma unroll
    for (int j = 0; j < 4; ++j) {
      int nb = n0 + (j << 4);
      if (nb < N) {
        v16bf b = load_frag(W, K, nb, k0);
        acc[j] = wmma_bf16(a, b, acc[j]);
      }
    }
  }
#pragma unroll
  for (int j = 0; j < 4; ++j) {
    int nb = n0 + (j << 4);
    if (nb >= N) continue;
    int n = nb + col;
    float bi = bias ? bias[n] : 0.f;
#pragma unroll
    for (int v = 0; v < 8; ++v) {
      int m = m0 + v + (hi << 3);
      float val = acc[j][v] + bi;
      if (EPI == 2) val = geluf(val);
      if (Cf)  Cf[(size_t)m * N + n] = val;
      if (Cbf) Cbf[(size_t)m * N + n] = f2bf(val);
    }
  }
}

// ---------------------------------------------------------------- causal conv
__global__ void __launch_bounds__(256)
conv_silu_k(const float* __restrict__ zx, const float* __restrict__ cw,
            const float* __restrict__ cb, float* __restrict__ xbcc) {
  int row = blockIdx.x, b = row / TT, t = row % TT, tid = threadIdx.x;
  for (int ch = tid; ch < CONVD; ch += 256) {
    float acc = cb[ch];
#pragma unroll
    for (int i = 0; i < 4; ++i) {
      int tt = t - 3 + i;
      if (tt >= 0) acc += zx[(size_t)(b * TT + tt) * DIP + DIN + ch] * cw[ch * 4 + i];
    }
    xbcc[(size_t)row * CONVD + ch] = siluf(acc);
  }
}

// ---------------------------------------------------------------- prep
__global__ void __launch_bounds__(256)
prep_k(const float* __restrict__ zx, const float* __restrict__ xbcc,
       const float* __restrict__ dt_bias, const float* __restrict__ A_log,
       __bf16* __restrict__ xdtT, __bf16* __restrict__ Bbf, __bf16* __restrict__ Cbf,
       float* __restrict__ adt) {
  int row = blockIdx.x, b = row / TT, t = row % TT, tid = threadIdx.x;
  __shared__ float s_dt[HH];
  if (tid < HH) {
    float d = softplusf(zx[(size_t)row * DIP + (DIN + CONVD) + tid] + dt_bias[tid]);
    s_dt[tid] = d;
    adt[(size_t)(b * HH + tid) * TT + t] = d * (-expf(A_log[tid]));
  }
  __syncthreads();
  for (int ch = tid; ch < DIN; ch += 256) {
    int h = ch >> 7, p = ch & 127;
    float v = xbcc[(size_t)row * CONVD + ch] * s_dt[h];
    xdtT[((size_t)(b * HH + h) * PP + p) * TT + t] = f2bf(v);
  }
  if (tid < NN)
    Bbf[(size_t)row * NN + tid] = f2bf(xbcc[(size_t)row * CONVD + DIN + tid]);
  else if (tid < 2 * NN) {
    int n = tid - NN;
    Cbf[(size_t)row * NN + n] = f2bf(xbcc[(size_t)row * CONVD + DIN + NN + n]);
  }
}

// ---------------------------------------------------------------- per-chunk cumsum
__global__ void __launch_bounds__(256)
acum_k(const float* __restrict__ adt, float* __restrict__ acum, float* __restrict__ asum) {
  int bh = blockIdx.x / NCH, c = blockIdx.x % NCH, k = threadIdx.x;
  __shared__ float s[CHK];
  s[k] = adt[(size_t)bh * TT + c * CHK + k];
  __syncthreads();
  for (int off = 1; off < CHK; off <<= 1) {
    float t2 = (k >= off) ? s[k - off] : 0.f;
    __syncthreads();
    s[k] += t2;
    __syncthreads();
  }
  acum[(size_t)bh * TT + c * CHK + k] = s[k];
  if (k == CHK - 1) asum[bh * NCH + c] = s[k];
}

// ---------------------------------------------------------------- SSD chunk
// per (b,c,h): A_cum tile staged via TDM (TENSORcnt); raw B tile staged via
// global_load_async_to_lds_b128 (ASYNCcnt) overlapped with phase-A WMMA;
// G=C*B^T (WMMA) -> masked exp -> LDS bf16 L; Y_diag=L*X (WMMA);
// states=(B*decay)^T*X (WMMA).  Dyn-LDS ~226KB (<320KB/WGP).
#define OFF_AC   0
#define OFF_M    1056
#define OFF_BD   (OFF_M + 256 * 256 * 2)
#define OFF_BRAW (OFF_BD + 64 * 256 * 2)
#define SSD_SMEM (OFF_BRAW + 64 * 256 * 4)
__global__ void __launch_bounds__(512)
ssd_chunk_k(const __bf16* __restrict__ Bbf, const __bf16* __restrict__ Cbf,
            const __bf16* __restrict__ xdtT, const float* __restrict__ xbcc,
            const float* __restrict__ acum, const float* __restrict__ asum,
            float* __restrict__ Y, float* __restrict__ states) {
  extern __shared__ char smraw[];
  float*  s_ac   = (float*)(smraw + OFF_AC);    // 256 + 1 floats
  __bf16* s_M    = (__bf16*)(smraw + OFF_M);    // 256 x 256
  __bf16* s_Bd   = (__bf16*)(smraw + OFF_BD);   // 64  x 256  [n][k]
  float*  s_Braw = (float*)(smraw + OFF_BRAW);  // 256 x 64   [k][n]
  int bid = blockIdx.x;
  int h = bid & 15, c = (bid >> 4) % NCH, b = bid / (16 * NCH);
  int tid = threadIdx.x, lane = tid & 31, w = tid >> 5;
  int col = lane & 15, hi = lane >> 4;
  size_t bh = (size_t)(b * HH + h);

  // --- stage A_cum row via Tensor Data Mover (one wave issues, EXEC ignored)
  if (tid == 0) {
    tdm_load_f32_row(acum + bh * TT + c * CHK, OFF_AC, CHK);
    s_ac[256] = asum[bh * NCH + c];
  }
  // --- stage raw B chunk (256 rows x 64 f32) via async global->LDS copies
  {
    const float* Bsrc = xbcc + (size_t)(b * TT + c * CHK) * CONVD + DIN;
#pragma unroll
    for (int j = 0; j < 8; ++j) {
      int idx = tid + (j << 9);              // 4096 x 16B chunks
      int k = idx >> 4, n4 = (idx & 15) << 2;
      const float* g = Bsrc + (size_t)k * CONVD + n4;
      unsigned l = OFF_BRAW + (unsigned)(((k << 6) + n4) << 2);
      asm volatile("global_load_async_to_lds_b128 %0, %1, off"
                   :: "v"(l), "v"(g) : "memory");
    }
  }
  __builtin_amdgcn_s_wait_tensorcnt(0);
  __syncthreads();                            // s_ac visible to all waves

  // ---- phase A: G = C * B^T, mask, exp -> s_M (wave w owns rows 16w..16w+15)
  int m0 = w << 4;
  const __bf16* Cb = Cbf + (size_t)(b * TT + c * CHK) * NN;
  const __bf16* Bb = Bbf + (size_t)(b * TT + c * CHK) * NN;
  v16bf a0 = load_frag(Cb, NN, m0, 0);
  v16bf a1 = load_frag(Cb, NN, m0, 32);
  for (int st = 0; st < 16; ++st) {
    v8f g = {};
    if (st <= w) {
      v16bf b0 = load_frag(Bb, NN, st << 4, 0);
      g = wmma_bf16(a0, b0, g);
      v16bf b1 = load_frag(Bb, NN, st << 4, 32);
      g = wmma_bf16(a1, b1, g);
    }
    int sIdx = (st << 4) + col;
#pragma unroll
    for (int v = 0; v < 8; ++v) {
      int k = m0 + v + (hi << 3);
      float val = (k >= sIdx) ? expf(s_ac[k] - s_ac[sIdx]) * g[v] : 0.f;
      s_M[k * 256 + sIdx] = f2bf(val);
    }
  }
  // ---- wait for our async copies, then build decayed-B bf16 from LDS
  asm volatile("s_wait_asynccnt 0x0" ::: "memory");
  __syncthreads();                            // everyone's copies + s_M done
  {
    float As = s_ac[256];
    for (int i = tid; i < NN * CHK; i += 512) {
      int n = i >> 8, k = i & 255;
      s_Bd[n * 256 + k] = f2bf(s_Braw[(k << 6) + n] * expf(As - s_ac[k]));
    }
  }
  __syncthreads();

  const __bf16* XT = xdtT + bh * PP * TT;
  // ---- phase C: Y_diag rows m0..m0+15 = s_M * X   (K=256, N=128)
  {
    v8f acc[8] = {};
    for (int kk = 0; kk < CHK; kk += 32) {
      v16bf afr = load_frag(s_M, 256, m0, kk);
#pragma unroll
      for (int j = 0; j < 8; ++j) {
        v16bf bfr = load_frag(XT, TT, j << 4, c * CHK + kk);
        acc[j] = wmma_bf16(afr, bfr, acc[j]);
      }
    }
    float* Yb = Y + (size_t)(b * TT + c * CHK) * DIN + h * PP;
#pragma unroll
    for (int j = 0; j < 8; ++j) {
      int p = (j << 4) + col;
#pragma unroll
      for (int v = 0; v < 8; ++v) {
        int k = m0 + v + (hi << 3);
        Yb[(size_t)k * DIN + p] = acc[j][v];
      }
    }
  }
  // ---- phase D: states[n,p] = s_Bd * X  (64x128, 2 tiles per wave)
  {
    float* Sb = states + ((bh * NCH + c) << 13);   // *8192
#pragma unroll
    for (int ti = 0; ti < 2; ++ti) {
      int tile = (w << 1) + ti;                    // 0..31
      int nt0 = (tile >> 3) << 4;
      int pt0 = (tile & 7) << 4;
      v8f acc = {};
      for (int kk = 0; kk < CHK; kk += 32) {
        v16bf afr = load_frag(s_Bd, 256, nt0, kk);
        v16bf bfr = load_frag(XT, TT, pt0, c * CHK + kk);
        acc = wmma_bf16(afr, bfr, acc);
      }
      int p = pt0 + col;
#pragma unroll
      for (int v = 0; v < 8; ++v) {
        int n = nt0 + v + (hi << 3);
        Sb[(size_t)p * NN + n] = acc[v];
      }
    }
  }
}

// ---------------------------------------------------------------- inter-chunk scan
__global__ void __launch_bounds__(256)
scan_k(const float* __restrict__ states, const float* __restrict__ asum,
       __bf16* __restrict__ sinbf) {
  int bh = blockIdx.x, tid = threadIdx.x;
  float S[32];
#pragma unroll
  for (int j = 0; j < 32; ++j) S[j] = 0.f;
  for (int c = 0; c < NCH; ++c) {
    size_t base = ((size_t)bh * NCH + c) << 13;
#pragma unroll
    for (int j = 0; j < 32; ++j) sinbf[base + tid + (j << 8)] = f2bf(S[j]);
    float d = expf(asum[bh * NCH + c]);
#pragma unroll
    for (int j = 0; j < 32; ++j) S[j] = S[j] * d + states[base + tid + (j << 8)];
  }
}

// ---------------------------------------------------------------- Y_off (WMMA)
__global__ void __launch_bounds__(256)
ssd_off_k(const __bf16* __restrict__ Cbf, const __bf16* __restrict__ sinbf,
          const float* __restrict__ acum, const float* __restrict__ xbcc,
          const float* __restrict__ Dskip, float* __restrict__ Y) {
  __shared__ float s_ac[CHK];
  int bid = blockIdx.x;
  int h = bid & 15, c = (bid >> 4) % NCH, b = bid / (16 * NCH);
  int tid = threadIdx.x, lane = tid & 31, w = tid >> 5;
  int col = lane & 15, hi = lane >> 4;
  size_t bh = (size_t)(b * HH + h);
  if (tid == 0) tdm_load_f32_row(acum + bh * TT + c * CHK, 0, CHK);
  __builtin_amdgcn_s_wait_tensorcnt(0);
  __syncthreads();
  float dsk = Dskip[h];
  const __bf16* Cb = Cbf + (size_t)(b * TT + c * CHK) * NN;
  const __bf16* Sb = sinbf + ((bh * NCH + c) << 13);
  for (int i = 0; i < 16; ++i) {
    int tile = w + (i << 3);          // 0..127
    int kt0 = (tile >> 3) << 4;
    int pt0 = (tile & 7) << 4;
    v8f acc = {};
    v16bf af = load_frag(Cb, NN, kt0, 0);
    v16bf bf = load_frag(Sb, NN, pt0, 0);
    acc = wmma_bf16(af, bf, acc);
    af = load_frag(Cb, NN, kt0, 32);
    bf = load_frag(Sb, NN, pt0, 32);
    acc = wmma_bf16(af, bf, acc);
    int p = pt0 + col;
#pragma unroll
    for (int v = 0; v < 8; ++v) {
      int k = kt0 + v + (hi << 3);
      size_t t = (size_t)(b * TT + c * CHK + k);
      float od = expf(s_ac[k]);
      float xs = xbcc[t * CONVD + h * PP + p];
      Y[t * DIN + h * PP + p] += od * acc[v] + dsk * xs;
    }
  }
}

// ---------------------------------------------------------------- gate + RMSNorm
__global__ void __launch_bounds__(256)
gate_rms_k(const float* __restrict__ Y, const float* __restrict__ zx,
           const float* __restrict__ nw, __bf16* __restrict__ ybf) {
  int row = blockIdx.x, tid = threadIdx.x;
  __shared__ float red[256];
  const float* Yr = Y + (size_t)row * DIN;
  const float* zr = zx + (size_t)row * DIP;
  float g[8], ss = 0.f;
#pragma unroll
  for (int j = 0; j < 8; ++j) {
    int ch = tid + (j << 8);
    float z = zr[ch];
    float v = Yr[ch] * siluf(z);
    g[j] = v; ss += v * v;
  }
  red[tid] = ss; __syncthreads();
  for (int off = 128; off; off >>= 1) { if (tid < off) red[tid] += red[tid + off]; __syncthreads(); }
  float inv = rsqrtf(red[0] * (1.0f / DIN) + 1e-5f);
#pragma unroll
  for (int j = 0; j < 8; ++j) {
    int ch = tid + (j << 8);
    ybf[(size_t)row * DIN + ch] = f2bf(g[j] * inv * nw[ch]);
  }
}

// ---------------------------------------------------------------- residual + LN3
__global__ void __launch_bounds__(256)
resid_ln_k(const float* __restrict__ mo, const float* __restrict__ localn,
           const float* __restrict__ w, const float* __restrict__ bvec,
           float* __restrict__ x3, __bf16* __restrict__ x3bf) {
  int row = blockIdx.x, b = row >> 11, l = row & 2047, tid = threadIdx.x;
  __shared__ float red[256];
  const float* mr = mo + (size_t)(b * TT + GG + l) * DM;
  const float* lr = localn + (size_t)row * DM;
  float vals[4], s = 0.f;
#pragma unroll
  for (int j = 0; j < 4; ++j) { int d = tid + (j << 8); vals[j] = mr[d] + lr[d]; s += vals[j]; }
  red[tid] = s; __syncthreads();
  for (int off = 128; off; off >>= 1) { if (tid < off) red[tid] += red[tid + off]; __syncthreads(); }
  float mu = red[0] * (1.0f / DM); __syncthreads();
  s = 0.f;
#pragma unroll
  for (int j = 0; j < 4; ++j) { float d = vals[j] - mu; s += d * d; }
  red[tid] = s; __syncthreads();
  for (int off = 128; off; off >>= 1) { if (tid < off) red[tid] += red[tid + off]; __syncthreads(); }
  float inv = rsqrtf(red[0] * (1.0f / DM) + 1e-5f);
#pragma unroll
  for (int j = 0; j < 4; ++j) {
    int d = tid + (j << 8);
    float y = (vals[j] - mu) * inv * w[d] + bvec[d];
    x3[(size_t)row * DM + d] = y;
    x3bf[(size_t)row * DM + d] = f2bf(y);
  }
}

// ---------------------------------------------------------------- final add
__global__ void __launch_bounds__(256)
add_k(const float* __restrict__ a, const float* __restrict__ b,
      float* __restrict__ o, int n) {
  int i = blockIdx.x * 256 + threadIdx.x;
  if (i < n) o[i] = a[i] + b[i];
}

// ================================================================ launch
extern "C" void kernel_launch(void* const* d_in, const int* in_sizes, int n_in,
                              void* d_out, int out_size, void* d_ws, size_t ws_size,
                              hipStream_t stream) {
  (void)in_sizes; (void)n_in; (void)out_size; (void)ws_size;
  const float* local_repr = (const float*)d_in[0];
  const float* global_repr = (const float*)d_in[1];
  const float* norm1_w = (const float*)d_in[2];
  const float* norm1_b = (const float*)d_in[3];
  const float* norm2_w = (const float*)d_in[4];
  const float* norm2_b = (const float*)d_in[5];
  const float* norm3_w = (const float*)d_in[6];
  const float* norm3_b = (const float*)d_in[7];
  const float* in_proj_w = (const float*)d_in[8];
  const float* conv_w = (const float*)d_in[9];
  const float* conv_b = (const float*)d_in[10];
  const float* dt_bias = (const float*)d_in[11];
  const float* A_log = (const float*)d_in[12];
  const float* D_skip = (const float*)d_in[13];
  const float* ssd_norm_w = (const float*)d_in[14];
  const float* out_proj_w = (const float*)d_in[15];
  const float* ffn_w1 = (const float*)d_in[16];
  const float* ffn_b1 = (const float*)d_in[17];
  const float* ffn_w2 = (const float*)d_in[18];
  const float* ffn_b2 = (const float*)d_in[19];
  float* out = (float*)d_out;

  char* ws = (char*)d_ws;
  size_t off = 0;
  auto alloc = [&](size_t bytes) -> char* {
    off = (off + 255) & ~(size_t)255;
    char* p = ws + off;
    off += bytes;
    return p;
  };
  __bf16* winbf  = (__bf16*)alloc((size_t)DIP * DM * 2);
  __bf16* woutbf = (__bf16*)alloc((size_t)DM * DIN * 2);
  __bf16* w1bf   = (__bf16*)alloc((size_t)(DM / 2) * DM * 2);
  __bf16* w2bf   = (__bf16*)alloc((size_t)DM * (DM / 2) * 2);
  __bf16* xnbf   = (__bf16*)alloc((size_t)BT * DM * 2);
  float*  localnf= (float*)alloc((size_t)BL * DM * 4);
  float*  zx     = (float*)alloc((size_t)BT * DIP * 4);
  float*  xbcc   = (float*)alloc((size_t)BT * CONVD * 4);
  __bf16* xdtT   = (__bf16*)alloc((size_t)BB * HH * PP * TT * 2);
  __bf16* Bbf    = (__bf16*)alloc((size_t)BT * NN * 2);
  __bf16* Cbf    = (__bf16*)alloc((size_t)BT * NN * 2);
  float*  adt    = (float*)alloc((size_t)BB * HH * TT * 4);
  float*  acum   = (float*)alloc((size_t)BB * HH * TT * 4);
  float*  asum   = (float*)alloc((size_t)BB * HH * NCH * 4);
  float*  Y      = (float*)alloc((size_t)BT * DIN * 4);
  float*  states = (float*)alloc((size_t)BB * HH * NCH * PP * NN * 4);
  __bf16* sinbf  = (__bf16*)alloc((size_t)BB * HH * NCH * PP * NN * 2);
  __bf16* ybf    = (__bf16*)alloc((size_t)BT * DIN * 2);
  float*  mo     = (float*)alloc((size_t)BT * DM * 4);
  float*  x3     = (float*)alloc((size_t)BL * DM * 4);
  __bf16* x3bf   = (__bf16*)alloc((size_t)BL * DM * 2);
  __bf16* hbf    = (__bf16*)alloc((size_t)BL * (DM / 2) * 2);
  float*  h2     = (float*)alloc((size_t)BL * DM * 4);

  auto cdiv = [](long a, long b) { return (int)((a + b - 1) / b); };
  auto gemm_blocks = [&](int M, int N) { return cdiv((long)(M >> 4) * ((N + 63) >> 6), 8); };

  // weights -> bf16
  tobf_k<<<cdiv((long)DIP * DM, 256), 256, 0, stream>>>(in_proj_w, winbf, DIP * DM);
  tobf_k<<<cdiv((long)DM * DIN, 256), 256, 0, stream>>>(out_proj_w, woutbf, DM * DIN);
  tobf_k<<<cdiv((long)(DM / 2) * DM, 256), 256, 0, stream>>>(ffn_w1, w1bf, (DM / 2) * DM);
  tobf_k<<<cdiv((long)DM * (DM / 2), 256), 256, 0, stream>>>(ffn_w2, w2bf, DM * (DM / 2));

  // LayerNorm global/local, fused concat + bf16 cast
  layernorm_k<<<BB * GG, 256, 0, stream>>>(global_repr, norm2_w, norm2_b,
                                           nullptr, xnbf, GG, 0);
  layernorm_k<<<BB * LL, 256, 0, stream>>>(local_repr, norm1_w, norm1_b,
                                           localnf, xnbf, LL, GG);

  // in_proj GEMM (9216 x 4240 x 1024)
  gemm_bf16_nt<0><<<gemm_blocks(BT, DIP), 256, 0, stream>>>(
      xnbf, winbf, nullptr, zx, nullptr, BT, DIP, DM);

  // conv + SiLU, prep, per-chunk cumsum
  conv_silu_k<<<BT, 256, 0, stream>>>(zx, conv_w, conv_b, xbcc);
  prep_k<<<BT, 256, 0, stream>>>(zx, xbcc, dt_bias, A_log, xdtT, Bbf, Cbf, adt);
  acum_k<<<BB * HH * NCH, 256, 0, stream>>>(adt, acum, asum);

  // SSD: diagonal + chunk states (WMMA + TDM + async-LDS, ~226KB dyn-LDS)
  ssd_chunk_k<<<BB * NCH * HH, 512, SSD_SMEM, stream>>>(
      Bbf, Cbf, xdtT, xbcc, acum, asum, Y, states);

  // inter-chunk state scan, then off-diagonal contribution
  scan_k<<<BB * HH, 256, 0, stream>>>(states, asum, sinbf);
  ssd_off_k<<<BB * NCH * HH, 256, 0, stream>>>(Cbf, sinbf, acum, xbcc, D_skip, Y);

  // gate + RMSNorm -> bf16, out_proj GEMM
  gate_rms_k<<<BT, 256, 0, stream>>>(Y, zx, ssd_norm_w, ybf);
  gemm_bf16_nt<0><<<gemm_blocks(BT, DM), 256, 0, stream>>>(
      ybf, woutbf, nullptr, mo, nullptr, BT, DM, DIN);

  // residual + LN3 -> f32 + bf16
  resid_ln_k<<<BL, 256, 0, stream>>>(mo, localnf, norm3_w, norm3_b, x3, x3bf);

  // FFN: GEMM1 (bias+GELU -> bf16), GEMM2 (bias -> f32), final residual
  gemm_bf16_nt<2><<<gemm_blocks(BL, DM / 2), 256, 0, stream>>>(
      x3bf, w1bf, ffn_b1, nullptr, hbf, BL, DM / 2, DM);
  gemm_bf16_nt<1><<<gemm_blocks(BL, DM), 256, 0, stream>>>(
      hbf, w2bf, ffn_b2, h2, nullptr, BL, DM, DM / 2);
  add_k<<<cdiv((long)BL * DM, 256), 256, 0, stream>>>(h2, x3, out, BL * DM);
}
